// GIN_31576599560634
// MI455X (gfx1250) — compile-verified, matrix operand order checked by reading, not run on hardware
//
#include <hip/hip_runtime.h>
#include <hip/hip_bf16.h>

// GIN forward for MI455X (gfx1250, wave32).
//  - Aggregation: fp32 global atomics (L2-resident; x = 51MB < 192MB L2)
//  - MLPs: bf16 activations/weights, fp32 accumulate via V_WMMA_F32_16X16X32_BF16
//  - Weights pre-transposed to [N][K] bf16 so A and B fragments are each two
//    contiguous 16B loads per lane (matches CDNA5 16-bit A-matrix VGPR layout).

typedef __attribute__((ext_vector_type(16))) __bf16 v16bf;
typedef __attribute__((ext_vector_type(8)))  __bf16 v8bf;
typedef __attribute__((ext_vector_type(8)))  float  v8f;

static const int  GN   = 50000;   // nodes
static const int  GE   = 800000;  // edges
static const long MPAD = 50176;   // 512 * 98, row padding for GEMM tiles

// ---- weight convert: WT[n][k] = bf16(W[k][n]) ----
__global__ __launch_bounds__(256)
void gin_wt_convert_k(const float* __restrict__ W, __bf16* __restrict__ WT,
                      int K, int Nout) {
  int idx = blockIdx.x * blockDim.x + threadIdx.x;
  if (idx >= K * Nout) return;
  int n = idx / K;
  int k = idx - n * K;
  WT[idx] = (__bf16)W[(long)k * Nout + n];
}

// ---- activation convert fp32 -> bf16 with zero padding rows ----
__global__ __launch_bounds__(256)
void gin_act_convert_k(const float* __restrict__ src, __bf16* __restrict__ dst,
                       long nReal) {
  long idx = (long)blockIdx.x * blockDim.x + threadIdx.x;  // < MPAD*256
  long row = idx >> 8;
  dst[idx] = (row < nReal) ? (__bf16)src[idx] : (__bf16)0.f;
}

// ---- edge aggregation: one wave per edge, 8 channels per lane ----
__global__ __launch_bounds__(256)
void gin_edge_agg_k(const float* __restrict__ x, const int* __restrict__ src,
                    const int* __restrict__ dst, float* __restrict__ hagg,
                    int E) {
  int t = blockIdx.x * blockDim.x + threadIdx.x;
  int e = t >> 5;
  if (e >= E) return;
  int c0 = (t & 31) << 3;
  long s = src[e];
  long d = dst[e];
  const float4* xp = (const float4*)(x + s * 256 + c0);
  float4 a = xp[0];
  float4 b = xp[1];
  float* hp = hagg + d * 256 + c0;
  atomicAdd(hp + 0, a.x); atomicAdd(hp + 1, a.y);
  atomicAdd(hp + 2, a.z); atomicAdd(hp + 3, a.w);
  atomicAdd(hp + 4, b.x); atomicAdd(hp + 5, b.y);
  atomicAdd(hp + 6, b.z); atomicAdd(hp + 7, b.w);
}

// ---- WMMA GEMM: out = act(A[MPAD][K] @ WT[N][K]^T + bias) ----
// Wave tile 64x64 (4x4 accumulator tiles), 8 waves/block.
// NOUT_T=256: waves 2(M)x4(N), block = 128 rows.  NOUT_T=64: 8(M)x1(N), 512 rows.
template <int NOUT_T, int K_T, bool RELU, bool OUTBF16>
__global__ __launch_bounds__(256)
void gin_gemm_wmma_k(const __bf16* __restrict__ A, const __bf16* __restrict__ WT,
                     const float* __restrict__ bias, void* __restrict__ out,
                     int nLimit) {
  constexpr int nWavesN = NOUT_T / 64;
  constexpr int nWavesM = 8 / nWavesN;
  const int lane  = threadIdx.x & 31;
  const int w     = threadIdx.x >> 5;
  const int waveN = w % nWavesN;
  const int waveM = w / nWavesN;
  const long row0 = (long)blockIdx.x * (64 * nWavesM) + waveM * 64;
  const int  col0 = waveN * 64;

  const int m     = lane & 15;          // row (A) / col (B) within 16-tile
  const int khalf = (lane >> 4) << 3;   // K half select per CDNA5 layout

  v8f acc[4][4] = {};

  for (int ks = 0; ks < K_T / 32; ++ks) {
    const int k0 = ks * 32 + khalf;
    v16bf afrag[4];
#pragma unroll
    for (int r = 0; r < 4; ++r) {
      const __bf16* p = A + (row0 + 16 * r + m) * (long)K_T + k0;
      v8bf lo = *(const v8bf*)p;        // K = k0 .. k0+7
      v8bf hi = *(const v8bf*)(p + 16); // K = k0+16 .. k0+23
#pragma unroll
      for (int i = 0; i < 8; ++i) { afrag[r][i] = lo[i]; afrag[r][8 + i] = hi[i]; }
    }
#pragma unroll
    for (int nt = 0; nt < 4; ++nt) {
      const __bf16* q = WT + (long)(col0 + 16 * nt + m) * K_T + k0;
      v8bf lo = *(const v8bf*)q;
      v8bf hi = *(const v8bf*)(q + 16);
      v16bf bfrag;
#pragma unroll
      for (int i = 0; i < 8; ++i) { bfrag[i] = lo[i]; bfrag[8 + i] = hi[i]; }
#pragma unroll
      for (int r = 0; r < 4; ++r) {
        acc[r][nt] = __builtin_amdgcn_wmma_f32_16x16x32_bf16(
            false, afrag[r], false, bfrag, (short)0, acc[r][nt], false, false);
      }
    }
  }

  // C/D layout: N = lane&15; M = vgpr + 8*(lane>=16)
  const int rowhalf = (lane >> 4) << 3;
#pragma unroll
  for (int nt = 0; nt < 4; ++nt) {
    const int col = col0 + 16 * nt + m;
    const float bv = bias[col];
#pragma unroll
    for (int r = 0; r < 4; ++r) {
      const long rbase = row0 + 16 * r + rowhalf;
#pragma unroll
      for (int vi = 0; vi < 8; ++vi) {
        float val = acc[r][nt][vi] + bv;
        if (RELU) val = val > 0.f ? val : 0.f;
        const long rr = rbase + vi;
        if (OUTBF16) {
          ((__bf16*)out)[rr * NOUT_T + col] = (__bf16)val;
        } else {
          if (rr < nLimit) ((float*)out)[rr * NOUT_T + col] = val;
        }
      }
    }
  }
}

extern "C" void kernel_launch(void* const* d_in, const int* in_sizes, int n_in,
                              void* d_out, int out_size, void* d_ws, size_t ws_size,
                              hipStream_t stream) {
  (void)in_sizes; (void)n_in; (void)out_size; (void)ws_size;

  const float* x_in = (const float*)d_in[0];
  const int*   src  = (const int*)d_in[1];
  const int*   dst  = (const int*)d_in[2];
  const float* W1s  = (const float*)d_in[3];
  const float* b1s  = (const float*)d_in[4];
  const float* W2s  = (const float*)d_in[5];
  const float* b2s  = (const float*)d_in[6];
  const float* W3s  = (const float*)d_in[7];
  const float* b3s  = (const float*)d_in[8];
  const float* cW1  = (const float*)d_in[9];
  const float* cb1  = (const float*)d_in[10];
  const float* cW2  = (const float*)d_in[11];
  const float* cb2  = (const float*)d_in[12];
  const float* cW3  = (const float*)d_in[13];
  const float* cb3  = (const float*)d_in[14];

  // ---- workspace carve (~155 MB) ----
  char* wsp = (char*)d_ws;
  auto carve = [&](size_t bytes) -> char* {
    char* p = wsp;
    wsp += (bytes + 255) & ~(size_t)255;
    return p;
  };
  float*  X    = (float*)carve((size_t)GN * 256 * 4);    // fp32 layer output
  float*  HAGG = (float*)carve((size_t)GN * 256 * 4);    // fp32 x + agg(x)
  __bf16* B1   = (__bf16*)carve((size_t)MPAD * 256 * 2); // bf16 activations
  __bf16* B2   = (__bf16*)carve((size_t)MPAD * 256 * 2);
  __bf16* WTi  = (__bf16*)carve((size_t)9 * 256 * 256 * 2);
  __bf16* WTc1 = (__bf16*)carve((size_t)256 * 256 * 2);
  __bf16* WTc2 = (__bf16*)carve((size_t)64  * 256 * 2);
  __bf16* WTc3 = (__bf16*)carve((size_t)256 * 64  * 2);

  // ---- weight bf16 transpose (tiny, once per launch) ----
  auto wconv = [&](const float* W, __bf16* WT, int K, int Nout) {
    int tot = K * Nout;
    gin_wt_convert_k<<<(tot + 255) / 256, 256, 0, stream>>>(W, WT, K, Nout);
  };
  for (int l = 0; l < 3; ++l) {
    wconv(W1s + (size_t)l * 65536, WTi + (size_t)(3 * l + 0) * 65536, 256, 256);
    wconv(W2s + (size_t)l * 65536, WTi + (size_t)(3 * l + 1) * 65536, 256, 256);
    wconv(W3s + (size_t)l * 65536, WTi + (size_t)(3 * l + 2) * 65536, 256, 256);
  }
  wconv(cW1, WTc1, 256, 256);
  wconv(cW2, WTc2, 256, 64);
  wconv(cW3, WTc3, 64, 256);

  const dim3 blk(256);
  const unsigned edgeBlocks = (unsigned)((GE * 32) / 256);
  const unsigned actBlocks  = (unsigned)(MPAD * 256 / 256);
  const unsigned g128 = (unsigned)(MPAD / 128);
  const unsigned g512 = (unsigned)(MPAD / 512);

  const float* cur = x_in;
  for (int l = 0; l < 3; ++l) {
    hipMemcpyAsync(HAGG, cur, (size_t)GN * 256 * 4, hipMemcpyDeviceToDevice, stream);
    gin_edge_agg_k<<<edgeBlocks, blk, 0, stream>>>(cur, src, dst, HAGG, GE);
    gin_act_convert_k<<<actBlocks, blk, 0, stream>>>(HAGG, B1, GN);
    gin_gemm_wmma_k<256, 256, true, true><<<g128, blk, 0, stream>>>(
        B1, WTi + (size_t)(3 * l + 0) * 65536, b1s + (size_t)l * 256, B2, 0);
    gin_gemm_wmma_k<256, 256, true, true><<<g128, blk, 0, stream>>>(
        B2, WTi + (size_t)(3 * l + 1) * 65536, b2s + (size_t)l * 256, B1, 0);
    gin_gemm_wmma_k<256, 256, false, false><<<g128, blk, 0, stream>>>(
        B1, WTi + (size_t)(3 * l + 2) * 65536, b3s + (size_t)l * 256, X, GN);
    cur = X;
  }

  // final conv: Dense(256,relu) -> Dense(64,relu) -> Dense(256,linear)
  hipMemcpyAsync(HAGG, cur, (size_t)GN * 256 * 4, hipMemcpyDeviceToDevice, stream);
  gin_edge_agg_k<<<edgeBlocks, blk, 0, stream>>>(cur, src, dst, HAGG, GE);
  gin_act_convert_k<<<actBlocks, blk, 0, stream>>>(HAGG, B1, GN);
  gin_gemm_wmma_k<256, 256, true, true><<<g128, blk, 0, stream>>>(B1, WTc1, cb1, B2, 0);
  gin_gemm_wmma_k<64, 256, true, true><<<g512, blk, 0, stream>>>(B2, WTc2, cb2, B1, 0);
  gin_gemm_wmma_k<256, 64, false, false><<<g128, blk, 0, stream>>>(B1, WTc3, cb3,
                                                                   (float*)d_out, GN);
}
